// MuARCLLM_13623636263001
// MI455X (gfx1250) — matrix operand order repaired and checked by reading, not compiled, not run on hardware
//
#include <hip/hip_runtime.h>

// ---------------- model constants ----------------
#define L_  4
#define D_  512
#define H_  2048
#define V_  32000
#define B_  2
#define T_  2048
#define BT_ 4096
#define EPS_ 1e-5f

typedef __bf16 bf16_t;
typedef __attribute__((ext_vector_type(16))) __bf16 v16bf;
typedef __attribute__((ext_vector_type(8)))  __bf16 v8bf;
typedef __attribute__((ext_vector_type(8)))  float  v8f;

// ---------------- epilogue flags ----------------
#define EPI_BIAS   1
#define EPI_GELU   2
#define EPI_RESID  4
#define EPI_BF16   8   // also store bf16 copy
#define EPI_NOF32 16   // skip f32 store (bf16 only)

// CDNA5 async global->LDS copy (ASYNCcnt path); 16 bytes per lane.
__device__ __forceinline__ void async_copy_b128(unsigned lds_off, const void* gaddr)
{
    asm volatile("global_load_async_to_lds_b128 %0, %1, off"
                 :: "v"(lds_off), "v"(gaddr) : "memory");
}

// =================================================================
// Tiled bf16 WMMA GEMM:  C(M,N) = A(M,K) * Bt(N,K)^T  (+bias/gelu/resid)
// A row-major MxK bf16 ; Bt row-major NxK bf16 (B stored transposed).
// Block = 256 threads = 8 waves; tile 128x64, K-step 32, double-buffered
// LDS fed by global_load_async_to_lds_b128 with prefetch distance 2.
// Waves 4(M) x 2(N); each wave 32x32 -> four 16x16 WMMA accumulators.
// =================================================================
template <int FLAGS>
__global__ __launch_bounds__(256) void gemm_wmma_bf16(
    const bf16_t* __restrict__ A,
    const bf16_t* __restrict__ Bt,
    const float*  __restrict__ bias,   // length N (or null)
    const float*  __restrict__ resid,  // MxN f32 (or null)
    float*        __restrict__ Cf,     // MxN f32 (or null if NOF32)
    bf16_t*       __restrict__ Cb,     // MxN bf16 (or null)
    int M, int N, int K,
    long long sA, long long sB, long long sC)
{
    const int bz = blockIdx.z;
    A  += (long long)bz * sA;
    Bt += (long long)bz * sB;
    const long long cbase = (long long)bz * sC;

    __shared__ __align__(16) bf16_t As[2][128][32 + 8];
    __shared__ __align__(16) bf16_t Bs[2][64][32 + 8];

    const int tid  = threadIdx.x;
    const int lane = tid & 31;
    const int wave = tid >> 5;      // 0..7
    const int wm   = wave >> 1;     // 0..3 : 32-row slice
    const int wn   = wave & 1;      // 0..1 : 32-col slice
    const int hf   = lane >> 4;     // half-wave select
    const int l15  = lane & 15;

    const int m0 = blockIdx.y * 128;
    const int n0 = blockIdx.x * 64;

    v8f acc00 = {}, acc01 = {}, acc10 = {}, acc11 = {};

    // ---- per-thread staging addresses (3 x b128 per thread per tile) ----
    const int srow = tid >> 2;            // 0..63
    const int skc  = (tid & 3) * 8;       // 0,8,16,24
    const bf16_t* ga0 = A  + (long long)(m0 + srow)      * K + skc;
    const bf16_t* ga1 = A  + (long long)(m0 + 64 + srow) * K + skc;
    const bf16_t* gb  = Bt + (long long)(n0 + srow)      * K + skc;
    unsigned la0[2], la1[2], lb[2];
#pragma unroll
    for (int b = 0; b < 2; ++b) {
        la0[b] = (unsigned)(size_t)&As[b][srow][skc];
        la1[b] = (unsigned)(size_t)&As[b][64 + srow][skc];
        lb[b]  = (unsigned)(size_t)&Bs[b][srow][skc];
    }

    const int nk = K >> 5;

    // ---- prologue: tiles 0 and 1 in flight ----
    async_copy_b128(la0[0], ga0);
    async_copy_b128(la1[0], ga1);
    async_copy_b128(lb[0],  gb);
    if (nk > 1) {
        async_copy_b128(la0[1], ga0 + 32);
        async_copy_b128(la1[1], ga1 + 32);
        async_copy_b128(lb[1],  gb  + 32);
    }

    for (int i = 0; i < nk; ++i) {
        const int cur = i & 1;
        // retire current tile, keep next tile's 3 async ops in flight
        if (i + 1 < nk) asm volatile("s_wait_asynccnt 3" ::: "memory");
        else            asm volatile("s_wait_asynccnt 0" ::: "memory");
        __syncthreads();

        // ---- fragments per documented CDNA5 VGPR layouts ----
        union { v16bf v; v8bf h[2]; } af0, af1, bfr0, bfr1;
        const int ar0 = wm * 32 + l15;
        af0.h[0] = *(const v8bf*)&As[cur][ar0][hf * 8];
        af0.h[1] = *(const v8bf*)&As[cur][ar0][16 + hf * 8];
        const int ar1 = ar0 + 16;
        af1.h[0] = *(const v8bf*)&As[cur][ar1][hf * 8];
        af1.h[1] = *(const v8bf*)&As[cur][ar1][16 + hf * 8];

        const int bc0 = wn * 32 + l15;
        bfr0.h[0] = *(const v8bf*)&Bs[cur][bc0][hf * 8];
        bfr0.h[1] = *(const v8bf*)&Bs[cur][bc0][16 + hf * 8];
        const int bc1 = bc0 + 16;
        bfr1.h[0] = *(const v8bf*)&Bs[cur][bc1][hf * 8];
        bfr1.h[1] = *(const v8bf*)&Bs[cur][bc1][16 + hf * 8];

        acc00 = __builtin_amdgcn_wmma_f32_16x16x32_bf16(
                    false, af0.v, false, bfr0.v, (short)0, acc00, false, false);
        acc01 = __builtin_amdgcn_wmma_f32_16x16x32_bf16(
                    false, af0.v, false, bfr1.v, (short)0, acc01, false, false);
        acc10 = __builtin_amdgcn_wmma_f32_16x16x32_bf16(
                    false, af1.v, false, bfr0.v, (short)0, acc10, false, false);
        acc11 = __builtin_amdgcn_wmma_f32_16x16x32_bf16(
                    false, af1.v, false, bfr1.v, (short)0, acc11, false, false);

        __syncthreads();   // all waves done reading buffer `cur`

        // refill the just-freed buffer with tile i+2
        if (i + 2 < nk) {
            const int k2 = (i + 2) << 5;
            async_copy_b128(la0[cur], ga0 + k2);
            async_copy_b128(la1[cur], ga1 + k2);
            async_copy_b128(lb[cur],  gb  + k2);
        }
        if (i + 3 < nk) {
            const int k3 = (i + 3) << 5;
            __builtin_prefetch(ga0 + k3, 0, 3);
            __builtin_prefetch(gb  + k3, 0, 3);
        }
    }

    // ---- epilogue: C layout VGPR i -> row hf*8+i, col l15 ----
#pragma unroll
    for (int r = 0; r < 2; ++r) {
        const v8f& aL = r ? acc10 : acc00;
        const v8f& aR = r ? acc11 : acc01;
#pragma unroll
        for (int i = 0; i < 8; ++i) {
            const int row  = m0 + wm * 32 + r * 16 + hf * 8 + i;
            const int col0 = n0 + wn * 32 + l15;
            const int col1 = col0 + 16;
            float v0 = aL[i];
            float v1 = aR[i];
            if (FLAGS & EPI_BIAS) { v0 += bias[col0]; v1 += bias[col1]; }
            if (FLAGS & EPI_GELU) {
                v0 = 0.5f * v0 * (1.0f + erff(v0 * 0.70710678118654752f));
                v1 = 0.5f * v1 * (1.0f + erff(v1 * 0.70710678118654752f));
            }
            const long long o0 = cbase + (long long)row * N + col0;
            const long long o1 = cbase + (long long)row * N + col1;
            if (FLAGS & EPI_RESID) { v0 += resid[o0]; v1 += resid[o1]; }
            if (!(FLAGS & EPI_NOF32)) { Cf[o0] = v0; Cf[o1] = v1; }
            if (FLAGS & (EPI_BF16 | EPI_NOF32)) {
                Cb[o0] = (bf16_t)v0; Cb[o1] = (bf16_t)v1;
            }
        }
    }
}

// =================================================================
// Transpose + convert fp32 (K x N) -> bf16 (N x K), batched over z
// =================================================================
__global__ __launch_bounds__(256) void cvt_transpose_bf16(
    const float* __restrict__ in, bf16_t* __restrict__ out, int K, int N)
{
    const long long base = (long long)blockIdx.z * K * N;
    const long long tot  = (long long)K * N;
    for (long long o = (long long)blockIdx.x * blockDim.x + threadIdx.x;
         o < tot; o += (long long)gridDim.x * blockDim.x) {
        const long long n = o / K;
        const long long k = o - n * K;
        out[base + o] = (bf16_t)in[base + k * N + n];
    }
}

// =================================================================
// Embedding gather: x[i,d] = embed[ids[i], d]
// =================================================================
__global__ __launch_bounds__(256) void gather_kernel(
    const int* __restrict__ ids, const float* __restrict__ embed,
    float* __restrict__ x)
{
    const long long o = (long long)blockIdx.x * blockDim.x + threadIdx.x;
    const long long i = o >> 9;          // / D_
    const long long d = o & (D_ - 1);
    x[o] = embed[(long long)ids[i] * D_ + d];
}

// =================================================================
// LayerNorm over D=512 (one block per row, 256 threads, 2 elems each)
// Optional outputs: fp32, bf16 row-major, bf16 transposed (per batch DxT)
// =================================================================
__global__ __launch_bounds__(256) void layernorm_kernel(
    const float* __restrict__ x, const float* __restrict__ sc,
    const float* __restrict__ bi, float* __restrict__ yf,
    bf16_t* __restrict__ yb, bf16_t* __restrict__ ybT)
{
    __shared__ float r1[256];
    __shared__ float r2[256];
    const int i   = blockIdx.x;           // 0..BT-1
    const int tid = threadIdx.x;
    const long long ro = (long long)i * D_;
    const float v0 = x[ro + tid];
    const float v1 = x[ro + tid + 256];
    r1[tid] = v0 + v1;
    r2[tid] = v0 * v0 + v1 * v1;
    __syncthreads();
    for (int st = 128; st > 0; st >>= 1) {
        if (tid < st) { r1[tid] += r1[tid + st]; r2[tid] += r2[tid + st]; }
        __syncthreads();
    }
    const float mu   = r1[0] * (1.0f / D_);
    const float var  = r2[0] * (1.0f / D_) - mu * mu;
    const float rstd = rsqrtf(var + EPS_);
    const float o0 = (v0 - mu) * rstd * sc[tid]       + bi[tid];
    const float o1 = (v1 - mu) * rstd * sc[tid + 256] + bi[tid + 256];
    if (yf) { yf[ro + tid] = o0; yf[ro + tid + 256] = o1; }
    if (yb) { yb[ro + tid] = (bf16_t)o0; yb[ro + tid + 256] = (bf16_t)o1; }
    if (ybT) {
        const int b = i >> 11;            // / T_
        const int t = i & (T_ - 1);
        ybT[(long long)b * D_ * T_ + (long long)tid * T_ + t]         = (bf16_t)o0;
        ybT[(long long)b * D_ * T_ + (long long)(tid + 256) * T_ + t] = (bf16_t)o1;
    }
}

// =================================================================
// Row dot product: d[i] = sum_k a[i,k]*b[i,k], K = 512
// =================================================================
__global__ __launch_bounds__(256) void rowdot_kernel(
    const float* __restrict__ a, const float* __restrict__ b,
    float* __restrict__ out)
{
    __shared__ float r[256];
    const int i = blockIdx.x, tid = threadIdx.x;
    const long long ro = (long long)i * D_;
    r[tid] = a[ro + tid] * b[ro + tid] + a[ro + tid + 256] * b[ro + tid + 256];
    __syncthreads();
    for (int st = 128; st > 0; st >>= 1) {
        if (tid < st) r[tid] += r[tid + st];
        __syncthreads();
    }
    if (tid == 0) out[i] = r[0];
}

// =================================================================
// Softmax over j of (2*s[t,j] - d[j])  == softmax(-dist^2) since
// s is symmetric (G is symmetric) and d_t is constant per row.
// One block per (b,t) row; T=2048 -> 8 elems per thread. Writes bf16.
// =================================================================
__global__ __launch_bounds__(256) void softmax_kernel(
    const float* __restrict__ s, const float* __restrict__ dvec,
    bf16_t* __restrict__ p)
{
    __shared__ float red[256];
    const int bt  = blockIdx.x;          // 0..B*T-1
    const int b   = bt >> 11;
    const int tid = threadIdx.x;
    const float* srow = s + (long long)bt * T_;
    const float* dv   = dvec + (long long)b * T_;

    float lg[8];
    float mx = -3.0e38f;
#pragma unroll
    for (int jj = 0; jj < 8; ++jj) {
        const int j = jj * 256 + tid;
        lg[jj] = 2.0f * srow[j] - dv[j];
        mx = fmaxf(mx, lg[jj]);
    }
    red[tid] = mx; __syncthreads();
    for (int st = 128; st > 0; st >>= 1) {
        if (tid < st) red[tid] = fmaxf(red[tid], red[tid + st]);
        __syncthreads();
    }
    const float m = red[0];
    __syncthreads();

    float sm = 0.0f;
#pragma unroll
    for (int jj = 0; jj < 8; ++jj) { lg[jj] = __expf(lg[jj] - m); sm += lg[jj]; }
    red[tid] = sm; __syncthreads();
    for (int st = 128; st > 0; st >>= 1) {
        if (tid < st) red[tid] += red[tid + st];
        __syncthreads();
    }
    const float inv = 1.0f / red[0];
#pragma unroll
    for (int jj = 0; jj < 8; ++jj) {
        const int j = jj * 256 + tid;
        p[(long long)bt * T_ + j] = (bf16_t)(lg[jj] * inv);
    }
}

// =================================================================
// Host orchestration
// =================================================================
extern "C" void kernel_launch(void* const* d_in, const int* in_sizes, int n_in,
                              void* d_out, int out_size, void* d_ws, size_t ws_size,
                              hipStream_t stream)
{
    (void)in_sizes; (void)n_in; (void)out_size; (void)ws_size;
    const int*   ids    = (const int*)  d_in[0];
    const float* embed  = (const float*)d_in[1];
    const float* g      = (const float*)d_in[2];
    const float* W1     = (const float*)d_in[3];
    const float* b1     = (const float*)d_in[4];
    const float* W2     = (const float*)d_in[5];
    const float* b2     = (const float*)d_in[6];
    const float* ln1_s  = (const float*)d_in[7];
    const float* ln1_b  = (const float*)d_in[8];
    const float* ln2_s  = (const float*)d_in[9];
    const float* ln2_b  = (const float*)d_in[10];
    const float* lnf_s  = (const float*)d_in[11];
    const float* lnf_b  = (const float*)d_in[12];
    const float* head_w = (const float*)d_in[13];
    const float* head_b = (const float*)d_in[14];
    float* out = (float*)d_out;

    // ---- carve workspace ----
    size_t off = 0;
    auto carve = [&](size_t bytes) -> char* {
        char* r = (char*)d_ws + off;
        off = (off + bytes + 255) & ~(size_t)255;
        return r;
    };
    float*  x     = (float*) carve((size_t)BT_ * D_ * 4);
    float*  y     = (float*) carve((size_t)BT_ * D_ * 4);
    float*  xg    = (float*) carve((size_t)BT_ * D_ * 4);
    float*  dvec  = (float*) carve((size_t)BT_ * 4);
    float*  sbuf  = (float*) carve((size_t)B_ * T_ * T_ * 4);
    bf16_t* ybf   = (bf16_t*)carve((size_t)BT_ * D_ * 2);
    bf16_t* ybfT  = (bf16_t*)carve((size_t)BT_ * D_ * 2);
    bf16_t* xgbf  = (bf16_t*)carve((size_t)BT_ * D_ * 2);
    bf16_t* y2bf  = (bf16_t*)carve((size_t)BT_ * D_ * 2);
    bf16_t* pbf   = (bf16_t*)carve((size_t)B_ * T_ * T_ * 2);
    bf16_t* hbf   = (bf16_t*)carve((size_t)BT_ * H_ * 2);
    bf16_t* gbf   = (bf16_t*)carve((size_t)L_ * D_ * D_ * 2);
    bf16_t* w1t   = (bf16_t*)carve((size_t)L_ * D_ * H_ * 2);
    bf16_t* w2t   = (bf16_t*)carve((size_t)L_ * D_ * H_ * 2);
    bf16_t* hdt   = (bf16_t*)carve((size_t)D_ * V_ * 2);

    // ---- weight conversion (transpose to NxK bf16) ----
    cvt_transpose_bf16<<<dim3(1024, 1, L_), 256, 0, stream>>>(g,      gbf, D_, D_);
    cvt_transpose_bf16<<<dim3(2048, 1, L_), 256, 0, stream>>>(W1,     w1t, D_, H_);
    cvt_transpose_bf16<<<dim3(2048, 1, L_), 256, 0, stream>>>(W2,     w2t, H_, D_);
    cvt_transpose_bf16<<<dim3(4096, 1, 1),  256, 0, stream>>>(head_w, hdt, D_, V_);

    // ---- embedding gather ----
    gather_kernel<<<(BT_ * D_) / 256, 256, 0, stream>>>(ids, embed, x);

    for (int l = 0; l < L_; ++l) {
        // ln1 -> y (f32), ybf, ybfT
        layernorm_kernel<<<BT_, 256, 0, stream>>>(
            x, ln1_s + l * D_, ln1_b + l * D_, y, ybf, ybfT);

        // xg = y @ g[l]   (f32 + bf16 outputs)
        gemm_wmma_bf16<EPI_BF16><<<dim3(D_ / 64, BT_ / 128, 1), 256, 0, stream>>>(
            ybf, gbf + (size_t)l * D_ * D_, nullptr, nullptr, xg, xgbf,
            BT_, D_, D_, 0, 0, 0);

        // d[i] = xg[i] . y[i]
        rowdot_kernel<<<BT_, 256, 0, stream>>>(xg, y, dvec);

        // s = xg @ y^T   (batched over B)
        gemm_wmma_bf16<0><<<dim3(T_ / 64, T_ / 128, B_), 256, 0, stream>>>(
            xgbf, ybf, nullptr, nullptr, sbuf, nullptr,
            T_, T_, D_,
            (long long)T_ * D_, (long long)T_ * D_, (long long)T_ * T_);

        // softmax(-dist^2) -> bf16 probabilities
        softmax_kernel<<<B_ * T_, 256, 0, stream>>>(sbuf, dvec, pbf);

        // x = x + probs @ y   (batched, fused residual)
        gemm_wmma_bf16<EPI_RESID><<<dim3(D_ / 64, T_ / 128, B_), 256, 0, stream>>>(
            pbf, ybfT, nullptr, x, x, nullptr,
            T_, D_, T_,
            (long long)T_ * T_, (long long)D_ * T_, (long long)T_ * D_);

        // ln2 -> y2bf only
        layernorm_kernel<<<BT_, 256, 0, stream>>>(
            x, ln2_s + l * D_, ln2_b + l * D_, nullptr, y2bf, nullptr);

        // h = gelu(y2 @ W1 + b1)  -> bf16 only
        gemm_wmma_bf16<EPI_BIAS | EPI_GELU | EPI_NOF32>
            <<<dim3(H_ / 64, BT_ / 128, 1), 256, 0, stream>>>(
            y2bf, w1t + (size_t)l * D_ * H_, b1 + l * H_, nullptr, nullptr, hbf,
            BT_, H_, D_, 0, 0, 0);

        // x = x + h @ W2 + b2
        gemm_wmma_bf16<EPI_BIAS | EPI_RESID>
            <<<dim3(D_ / 64, BT_ / 128, 1), 256, 0, stream>>>(
            hbf, w2t + (size_t)l * D_ * H_, b2 + l * D_, x, x, nullptr,
            BT_, D_, H_, 0, 0, 0);
    }

    // final LN -> reuse y2bf
    layernorm_kernel<<<BT_, 256, 0, stream>>>(x, lnf_s, lnf_b, nullptr, y2bf, nullptr);

    // logits = yF @ head_w + head_b  -> d_out (f32)
    gemm_wmma_bf16<EPI_BIAS><<<dim3(V_ / 64, BT_ / 128, 1), 256, 0, stream>>>(
        y2bf, hdt, head_b, nullptr, out, nullptr,
        BT_, V_, D_, 0, 0, 0);
}
